// ContextualEncoder_68298569941193
// MI455X (gfx1250) — matrix operand order repaired and checked by reading, not compiled, not run on hardware
//
#include <hip/hip_runtime.h>
#include <math.h>

// ---------------------------------------------------------------------------
// ContextualEncoder for MI455X (gfx1250): bf16 WMMA GEMMs + persistent
// LDS-resident LSTM recurrence. Wh staging uses the Tensor Data Mover
// (tensor_load_to_lds + s_wait_tensorcnt). Per-step sync: L2 phase barrier.
// A=4, B=8, S=2048, D=256, L=3 (2 layer iterations). All inputs f32.
// ---------------------------------------------------------------------------

typedef __attribute__((ext_vector_type(16))) __bf16       v16bf;
typedef __attribute__((ext_vector_type(8)))  float        v8f;
typedef __attribute__((ext_vector_type(4)))  unsigned int u32x4;
typedef __attribute__((ext_vector_type(8)))  int          i32x8;
typedef __attribute__((ext_vector_type(4)))  int          i32x4;

#define AA 4
#define BB 8
#define SS 2048
#define DD 256
#define RR (AA * BB * SS)          // 65536 activation rows
#define NB (AA * BB)               // 32 batch rows in the recurrence

__device__ inline float sigmoidf_(float x) { return 1.0f / (1.0f + __expf(-x)); }

__device__ inline v8f wmma_bf16(v16bf a, v16bf b, v8f c) {
  return __builtin_amdgcn_wmma_f32_16x16x32_bf16(
      /*neg_a=*/false, a, /*neg_b=*/false, b,
      /*c_mod=*/(short)0, c, /*reuse_a=*/false, /*reuse_b=*/false);
}

// A fragment (16x32 bf16, MxK) from row-major bf16 [M x lda].
// ISA layout: lanes 0-15 hold M=lane, elems = K{k0..k0+7, k0+16..k0+23};
//             lanes 16-31 hold M=lane-16, elems = K{k0+8..k0+15, k0+24..k0+31}.
__device__ inline v16bf load_afrag(const __bf16* __restrict__ A, int lda,
                                   int m0, int k0) {
  int lane = threadIdx.x & 31;
  int m  = m0 + (lane & 15);
  int kb = k0 + ((lane >> 4) << 3);
  const __bf16* p = A + (size_t)m * lda + kb;
  union { v16bf v; uint4 q[2]; } u;
  u.q[0] = *(const uint4*)(p);        // K kb .. kb+7
  u.q[1] = *(const uint4*)(p + 16);   // K kb+16 .. kb+23
  return u.v;
}

// B fragment from pre-packed weights: [nTile][kTile][lane][16] bf16.
__device__ inline v16bf load_bfrag(const __bf16* __restrict__ Bp,
                                   int nTile, int kTile, int kTiles) {
  int lane = threadIdx.x & 31;
  return ((const v16bf*)Bp)[((size_t)(nTile * kTiles + kTile)) * 32 + lane];
}

// ---------------------------------------------------------------------------
// Weight packer: W f32 [K x N] row-major -> bf16 B-fragments,
// layout [nTile][kTile][lane][16].
// B frag (16x16x32): lane<16: N=nT*16+lane, elems=K kT*32+{0..7,16..23};
//                    lane>=16: N=nT*16+lane-16, elems=K kT*32+{8..15,24..31}.
// ---------------------------------------------------------------------------
__global__ __launch_bounds__(256) void pack_b_kernel(const float* __restrict__ W,
                                                     __bf16* __restrict__ out,
                                                     int K, int N) {
  int idx = blockIdx.x * 256 + threadIdx.x;
  if (idx >= K * N) return;
  int e    = idx & 15;
  int lane = (idx >> 4) & 31;
  int t    = idx >> 9;
  int kTiles = K >> 5;
  int kT = t % kTiles, nT = t / kTiles;
  int n = nT * 16 + (lane & 15);
  int k = kT * 32 + ((lane >> 4) << 3) + (e & 7) + ((e >> 3) << 4);
  out[idx] = (__bf16)W[(size_t)k * N + n];
}

// Wh packer: slice-major so each WG's 128 KB is ONE contiguous block that a
// single TDM descriptor can move: [slice][lt = g*4+c2][kTile][lane][16],
// where global nTile = g*16 + slice*4 + c2 (gate g, col-tile c2).
__global__ __launch_bounds__(256) void pack_wh_kernel(const float* __restrict__ W,
                                                      __bf16* __restrict__ out) {
  const int K = DD, N = 4 * DD;
  int idx = blockIdx.x * 256 + threadIdx.x;
  if (idx >= K * N) return;
  int e    = idx & 15;
  int lane = (idx >> 4) & 31;
  int r    = idx >> 9;                 // (slice,lt,kT) flattened
  int kT   = r & 7;
  int q    = r >> 3;                   // 0..63
  int slice = q >> 4;
  int lt    = q & 15;
  int g = lt >> 2, c2 = lt & 3;
  int nT = g * 16 + slice * 4 + c2;
  int n = nT * 16 + (lane & 15);
  int k = kT * 32 + ((lane >> 4) << 3) + (e & 7) + ((e >> 3) << 4);
  out[idx] = (__bf16)W[(size_t)k * N + n];
}

__global__ __launch_bounds__(256) void cvt_bf16_kernel(const float* __restrict__ in,
                                                       __bf16* __restrict__ out, int n) {
  int i = blockIdx.x * 256 + threadIdx.x;
  if (i < n) out[i] = (__bf16)in[i];
}

__global__ void init_sync_kernel(int* s) {
  if (threadIdx.x < 8) s[threadIdx.x] = 0;
}

// ---------------------------------------------------------------------------
// zp = (sum_a h[a,b,S-1,:] / (A-1)) @ W4 + b4     -> zp[B x D], one workgroup.
// ---------------------------------------------------------------------------
__global__ __launch_bounds__(256) void zp_kernel(const float* __restrict__ h,
                                                 const float* __restrict__ W4,
                                                 const float* __restrict__ b4,
                                                 float* __restrict__ zp) {
  __shared__ float z[BB * DD];
  int tid = threadIdx.x;                       // = d
  for (int b = 0; b < BB; ++b) {
    float s = 0.f;
    for (int a = 0; a < AA; ++a)
      s += h[(((size_t)a * BB + b) * SS + (SS - 1)) * DD + tid];
    z[b * DD + tid] = s * (1.0f / (AA - 1));
  }
  __syncthreads();
  for (int b = 0; b < BB; ++b) {
    float acc = b4[tid];
    for (int k = 0; k < DD; ++k)
      acc += z[b * DD + k] * W4[(size_t)k * DD + tid];
    zp[b * DD + tid] = acc;
  }
}

// ---------------------------------------------------------------------------
// Generic WMMA GEMM: C[M,N] = act(A_bf16[M,K] @ Bpacked + bias (+ zp[b,n]))
// Each wave: one 16-row M-tile x 64 cols (4 N-tiles). Rows map a*B*S+b*S+s,
// so a 16-row tile has constant b (S multiple of 16) for the zp broadcast.
// ---------------------------------------------------------------------------
template <bool TANH, bool OUTBF>
__global__ __launch_bounds__(256) void gemm_wmma(const __bf16* __restrict__ A,
                                                 const __bf16* __restrict__ Bp,
                                                 const float* __restrict__ bias,
                                                 const float* __restrict__ zp,
                                                 float* __restrict__ outF,
                                                 __bf16* __restrict__ outB,
                                                 int M, int N, int K) {
  int kTiles = K >> 5;
  int NG     = N >> 6;
  int wid    = blockIdx.x * 8 + (threadIdx.x >> 5);
  int mTile  = wid / NG;
  int ng     = wid % NG;
  if (mTile >= (M >> 4)) return;
  int m0 = mTile << 4;
  int n0 = ng << 6;
  int lane = threadIdx.x & 31;

  v8f acc0 = {}, acc1 = {}, acc2 = {}, acc3 = {};
  int nT0 = n0 >> 4;
  for (int kt = 0; kt < kTiles; ++kt) {
    v16bf a  = load_afrag(A, K, m0, kt << 5);
    v16bf b0 = load_bfrag(Bp, nT0 + 0, kt, kTiles);
    v16bf b1 = load_bfrag(Bp, nT0 + 1, kt, kTiles);
    v16bf b2 = load_bfrag(Bp, nT0 + 2, kt, kTiles);
    v16bf b3 = load_bfrag(Bp, nT0 + 3, kt, kTiles);
    acc0 = wmma_bf16(a, b0, acc0);
    acc1 = wmma_bf16(a, b1, acc1);
    acc2 = wmma_bf16(a, b2, acc2);
    acc3 = wmma_bf16(a, b3, acc3);
  }

  int nl   = lane & 15;
  int mb   = m0 + ((lane >> 4) << 3);
  int bidx = (m0 >> 11) & (BB - 1);           // (row/S) % B

  auto epi = [&](v8f acc, int j) {
    int n = n0 + (j << 4) + nl;
    float badd = bias ? bias[n] : 0.0f;
    if (zp) badd += zp[bidx * DD + n];
#pragma unroll
    for (int r = 0; r < 8; ++r) {
      float v = acc[r] + badd;
      if (TANH) v = tanhf(v);
      size_t o = (size_t)(mb + r) * N + n;
      if (OUTBF) outB[o] = (__bf16)v;
      else       outF[o] = v;
    }
  };
  epi(acc0, 0); epi(acc1, 1); epi(acc2, 2); epi(acc3, 3);
}

// ---------------------------------------------------------------------------
// Bidirectional LSTM recurrence. Grid = 8 WGs of 256 threads:
//   blockIdx = dir*4 + slice.  Each WG owns a 64-wide h-slice of one
// direction; its 128 KB of Wh fragments are moved into LDS by ONE Tensor
// Data Mover descriptor (1D tile, data_size=8B, 16384 elements), c stays in
// VGPRs, and each step computes gates = h_prev @ Wh_slice + xw via WMMA,
// then the elementwise update. h_t (bf16) goes into y[:, dir*256+slice*64..],
// which doubles as the next step's A operand and the final GEMM input.
// Per-direction step sync: L2 atomic phase barrier among 4 WGs.
// ---------------------------------------------------------------------------
__global__ __launch_bounds__(256) void lstm_wmma(const float* __restrict__ xw_f,
                                                 const float* __restrict__ xw_b,
                                                 const __bf16* __restrict__ WhP_f,
                                                 const __bf16* __restrict__ WhP_b,
                                                 __bf16* __restrict__ y,
                                                 int* __restrict__ sync,
                                                 int layer) {
  extern __shared__ __bf16 lds[];              // 16 tiles * 8 kt * 512 bf16 = 128 KB
  int dir   = blockIdx.x >> 2;
  int slice = blockIdx.x & 3;
  const float*  __restrict__ xw  = dir ? xw_b  : xw_f;
  const __bf16* __restrict__ WhP = dir ? WhP_b : WhP_f;

  int tid  = threadIdx.x;
  int lane = tid & 31;
  int wave = tid >> 5;
  int mt   = wave & 1;
  int ct   = wave >> 1;

  // ---- Stage this slice's Wh fragments (contiguous 128 KB) via TDM. ----
  const __bf16* src = WhP + (size_t)slice * (16 * 8 * 512);
#if __has_builtin(__builtin_amdgcn_tensor_load_to_lds)
  if (wave == 0) {
    unsigned long long gaddr = (unsigned long long)(uintptr_t)src;
    unsigned int ldsAddr = (unsigned int)(uintptr_t)&lds[0];  // low 32b = LDS byte offset
    // D# group 0: count=1 | lds_addr | global_addr[56:0] | type=2
    u32x4 g0;
    g0.x = 1u;
    g0.y = ldsAddr;
    g0.z = (unsigned int)(gaddr & 0xffffffffu);
    g0.w = (unsigned int)((gaddr >> 32) & 0x1ffffffu) | (2u << 30);
    // D# group 1: data_size=3 (8B); tensor_dim0=tile_dim0=16384 elems (=128KB);
    // tensor_dim1=1; tile_dim1/2=0 (unused); dim0 stride = 16384.
    i32x8 g1;
    g1[0] = (3 << 16);                  // wg_mask=0, data_size=8B
    g1[1] = (int)(16384u << 16);        // tensor_dim0[15:0] @ bits[63:48]
    g1[2] = (int)(1u << 16);            // tensor_dim0[31:16]=0, tensor_dim1[15:0]=1
    g1[3] = (int)(16384u << 16);        // tensor_dim1[31:16]=0, tile_dim0=16384
    g1[4] = 0;                          // tile_dim1=0, tile_dim2=0
    g1[5] = 16384;                      // tensor_dim0_stride[31:0]
    g1[6] = 0;
    g1[7] = 0;
    i32x4 z4 = {0, 0, 0, 0};
#if defined(__clang_major__) && (__clang_major__ >= 23)
    i32x8 z8 = {0, 0, 0, 0, 0, 0, 0, 0};
    __builtin_amdgcn_tensor_load_to_lds(g0, g1, z4, z4, z8, 0);
#else
    __builtin_amdgcn_tensor_load_to_lds(g0, g1, z4, z4, 0);
#endif
#if __has_builtin(__builtin_amdgcn_s_wait_tensorcnt)
    __builtin_amdgcn_s_wait_tensorcnt(0);
#else
    asm volatile("s_wait_tensorcnt 0x0" ::: "memory");
#endif
  }
#else
  { // Fallback: manual staging (slice block is contiguous).
    const uint4* Wsrc = (const uint4*)src;
    uint4*       Wdst = (uint4*)lds;
    for (int i = tid; i < 16 * 512; i += 256) Wdst[i] = Wsrc[i];
  }
#endif
  __syncthreads();

  int* cnt   = sync + dir * 2;
  int* phase = sync + dir * 2 + 1;

  float c[8];
#pragma unroll
  for (int r = 0; r < 8; ++r) c[r] = 0.0f;

  int nl      = lane & 15;
  int mb      = mt * 16 + ((lane >> 4) << 3);       // D-frag base batch row
  int colBase = slice * 64 + ct * 16 + nl;          // col in 256-wide h space
  const v16bf* ldsF = (const v16bf*)lds;

  for (int t = 0; t < SS; ++t) {
    int s = dir ? (SS - 1 - t) : t;
    v8f gi = {}, gf = {}, gc = {}, go = {};

    if (t > 0) {
      int sPrev = dir ? (s + 1) : (s - 1);
#pragma unroll
      for (int kt = 0; kt < 8; ++kt) {
        // A fragment of h_prev: row = batch n, K = this direction's 256 cols.
        int m  = mt * 16 + nl;
        int kb = (kt << 5) + ((lane >> 4) << 3);
        const __bf16* p = y + ((size_t)m * SS + sPrev) * 512 + dir * 256 + kb;
        union { v16bf v; uint4 q[2]; } u;
        u.q[0] = *(const uint4*)(p);
        u.q[1] = *(const uint4*)(p + 16);
        v16bf a = u.v;
        gi = wmma_bf16(a, ldsF[((0 * 4 + ct) * 8 + kt) * 32 + lane], gi);
        gf = wmma_bf16(a, ldsF[((1 * 4 + ct) * 8 + kt) * 32 + lane], gf);
        gc = wmma_bf16(a, ldsF[((2 * 4 + ct) * 8 + kt) * 32 + lane], gc);
        go = wmma_bf16(a, ldsF[((3 * 4 + ct) * 8 + kt) * 32 + lane], go);
      }
    }

#pragma unroll
    for (int r = 0; r < 8; ++r) {
      int nb = mb + r;                                   // batch row 0..31
      size_t rowOff = ((size_t)nb * SS + s) * (4 * DD);
      float xi = xw[rowOff + 0 * DD + colBase];
      float xf = xw[rowOff + 1 * DD + colBase];
      float xg = xw[rowOff + 2 * DD + colBase];
      float xo = xw[rowOff + 3 * DD + colBase];
      float iv = sigmoidf_(gi[r] + xi);
      float fv = sigmoidf_(gf[r] + xf);
      float gv = tanhf(gc[r] + xg);
      float ov = sigmoidf_(go[r] + xo);
      float cv = fv * c[r] + iv * gv;
      c[r] = cv;
      float hv = ov * tanhf(cv);
      y[((size_t)nb * SS + s) * 512 + dir * 256 + colBase] = (__bf16)hv;
      if (t + 1 < SS) {                                  // warm L2 for next step
        int s2 = dir ? (s - 1) : (s + 1);
        __builtin_prefetch(xw + ((size_t)nb * SS + s2) * (4 * DD) + colBase, 0, 1);
      }
    }

    __threadfence();                                     // release h_t to L2
    __syncthreads();
    if (tid == 0) {
      int target = layer * SS + t + 1;                   // monotone phase
      int prev = __hip_atomic_fetch_add(cnt, 1, __ATOMIC_ACQ_REL,
                                        __HIP_MEMORY_SCOPE_AGENT);
      if (prev == 3) {
        __hip_atomic_store(cnt, 0, __ATOMIC_RELAXED, __HIP_MEMORY_SCOPE_AGENT);
        __hip_atomic_fetch_add(phase, 1, __ATOMIC_RELEASE,
                               __HIP_MEMORY_SCOPE_AGENT);
      } else {
        while (__hip_atomic_load(phase, __ATOMIC_ACQUIRE,
                                 __HIP_MEMORY_SCOPE_AGENT) < target) {
          __builtin_amdgcn_s_sleep(1);
        }
      }
    }
    __syncthreads();
  }
}

// ---------------------------------------------------------------------------
// Host-side orchestration (all on `stream`, graph-capture safe).
// ---------------------------------------------------------------------------
extern "C" void kernel_launch(void* const* d_in, const int* in_sizes, int n_in,
                              void* d_out, int out_size, void* d_ws, size_t ws_size,
                              hipStream_t stream) {
  (void)in_sizes; (void)n_in; (void)out_size; (void)ws_size;

  const float* x    = (const float*)d_in[0];
  const float* W3   = (const float*)d_in[1];
  const float* b3   = (const float*)d_in[2];
  const float* W4   = (const float*)d_in[3];
  const float* b4   = (const float*)d_in[4];
  const float* Wx_f = (const float*)d_in[5];
  const float* Wh_f = (const float*)d_in[6];
  const float* b_f  = (const float*)d_in[7];
  const float* Wx_b = (const float*)d_in[8];
  const float* Wh_b = (const float*)d_in[9];
  const float* b_b  = (const float*)d_in[10];
  const float* Wd   = (const float*)d_in[11];
  const float* bd   = (const float*)d_in[12];
  float* out = (float*)d_out;

  char* ws = (char*)d_ws;
  size_t off = 0;
  auto alloc = [&](size_t bytes) -> char* {
    char* p = ws + off;
    off += (bytes + 255) & ~(size_t)255;
    return p;
  };

  int*    syncBuf = (int*)   alloc(64);
  float*  zp      = (float*) alloc((size_t)BB * DD * 4);
  __bf16* W3p     = (__bf16*)alloc((size_t)DD * DD * 2);
  __bf16* Wxfp    = (__bf16*)alloc((size_t)DD * 4 * DD * 2);
  __bf16* Whfp    = (__bf16*)alloc((size_t)DD * 4 * DD * 2);
  __bf16* Wxbp    = (__bf16*)alloc((size_t)DD * 4 * DD * 2);
  __bf16* Whbp    = (__bf16*)alloc((size_t)DD * 4 * DD * 2);
  __bf16* Wdp     = (__bf16*)alloc((size_t)2 * DD * DD * 2);
  __bf16* hbf     = (__bf16*)alloc((size_t)RR * DD * 2);
  __bf16* fbf     = (__bf16*)alloc((size_t)RR * DD * 2);
  __bf16* ybf     = (__bf16*)alloc((size_t)RR * 2 * DD * 2);
  float*  xwf     = (float*) alloc((size_t)RR * 4 * DD * 4);
  float*  xwb     = (float*) alloc((size_t)RR * 4 * DD * 4);
  float*  hbuf    = (float*) alloc((size_t)RR * DD * 4);

  init_sync_kernel<<<1, 64, 0, stream>>>(syncBuf);

  // Pack weights into WMMA B-fragment order (bf16). Wh gets the
  // slice-contiguous layout consumed by the TDM descriptor.
  pack_b_kernel<<<(DD * DD) / 256, 256, 0, stream>>>(W3, W3p, DD, DD);
  pack_b_kernel<<<(DD * 4 * DD) / 256, 256, 0, stream>>>(Wx_f, Wxfp, DD, 4 * DD);
  pack_wh_kernel<<<(DD * 4 * DD) / 256, 256, 0, stream>>>(Wh_f, Whfp);
  pack_b_kernel<<<(DD * 4 * DD) / 256, 256, 0, stream>>>(Wx_b, Wxbp, DD, 4 * DD);
  pack_wh_kernel<<<(DD * 4 * DD) / 256, 256, 0, stream>>>(Wh_b, Whbp);
  pack_b_kernel<<<(2 * DD * DD) / 256, 256, 0, stream>>>(Wd, Wdp, 2 * DD, DD);

  const int nElems = RR * DD;                  // 16.7M activations
  for (int layer = 0; layer < 2; ++layer) {
    const float* hsrc = (layer == 0) ? x : hbuf;
    float* hdst = (layer == 0) ? hbuf : out;

    // zp = (sum_a h[...,-1,:]/(A-1)) @ W4 + b4
    zp_kernel<<<1, 256, 0, stream>>>(hsrc, W4, b4, zp);

    // h -> bf16 for GEMM A operand
    cvt_bf16_kernel<<<nElems / 256, 256, 0, stream>>>(hsrc, hbf, nElems);

    // f = tanh(h @ W3 + b3 + zp)                 [65536 x 256] bf16
    gemm_wmma<true, true><<<(RR / 16) * (DD / 64) / 8, 256, 0, stream>>>(
        hbf, W3p, b3, zp, nullptr, fbf, RR, DD, DD);

    // xw_{f,b} = f @ Wx_{f,b} + b_{f,b}          [65536 x 1024] f32
    gemm_wmma<false, false><<<(RR / 16) * (4 * DD / 64) / 8, 256, 0, stream>>>(
        fbf, Wxfp, b_f, nullptr, xwf, nullptr, RR, 4 * DD, DD);
    gemm_wmma<false, false><<<(RR / 16) * (4 * DD / 64) / 8, 256, 0, stream>>>(
        fbf, Wxbp, b_b, nullptr, xwb, nullptr, RR, 4 * DD, DD);

    // Bidirectional recurrence -> y = [fwd|bwd]  [65536 x 512] bf16
    lstm_wmma<<<8, 256, 131072, stream>>>(xwf, xwb, Whfp, Whbp, ybf, syncBuf, layer);

    // h_next = y @ Wd + bd                       [65536 x 256] f32
    gemm_wmma<false, false><<<(RR / 16) * (DD / 64) / 8, 256, 0, stream>>>(
        ybf, Wdp, bd, nullptr, hdst, nullptr, RR, DD, 2 * DD);
  }
}
